// TJDLayer_18605798326574
// MI455X (gfx1250) — matrix operand order repaired and checked by reading, not compiled, not run on hardware
//
#include <hip/hip_runtime.h>
#include <math.h>

#define AS1 __attribute__((address_space(1)))
#define AS3 __attribute__((address_space(3)))

typedef __attribute__((ext_vector_type(16))) __bf16 v16bf;
typedef __attribute__((ext_vector_type(8)))  float  v8f;
typedef int v4i_native __attribute__((vector_size(16)));   // matches builtin param type

union Frag32B {
    struct { uint4 lo, hi; } u;
    v16bf v;
};

#if defined(__gfx1250__) && __has_builtin(__builtin_amdgcn_global_load_async_to_lds_b128)
#define USE_ASYNC 1
#else
#define USE_ASYNC 0
#endif

__device__ __forceinline__ unsigned cvt_bf16(float f) {
    unsigned u = __float_as_uint(f);
    return (u + 0x7fffu + ((u >> 16) & 1u)) >> 16;   // round-to-nearest-even
}
__device__ __forceinline__ unsigned pack_bf16(float lo, float hi) {
    return cvt_bf16(lo) | (cvt_bf16(hi) << 16);
}

// ---------------------------------------------------------------- phase 0: xbar = mean_t(x)
__global__ void k_xbar(const float* __restrict__ x, float* __restrict__ xbar) {
    int b = blockIdx.x;                       // 32 blocks
    for (int d = threadIdx.x; d < 1024; d += blockDim.x) {
        const float* p = x + (size_t)b * 14 * 1024 + d;
        float s = 0.f;
        #pragma unroll
        for (int t = 0; t < 14; ++t) s += p[t * 1024];
        xbar[b * 1024 + d] = s * (1.0f / 14.0f);
    }
}

// ---------------------------------------------------------------- phase 0: A-fragment swizzle (bf16)
// Layout: flat g = (kc*2+mt)*32 + lane; 16 bf16 per (g): VGPR0..3 = K=kh*8+{0..7},
// VGPR4..7 = K=16+kh*8+{0..7}; M = lane&15 (per CDNA5 16-bit A 16x32 layout).
__global__ void k_swizzleA(const float* __restrict__ xbar, unsigned short* __restrict__ A) {
    int g = blockIdx.x * blockDim.x + threadIdx.x;  // 0..2047
    if (g >= 2048) return;
    int kc = g >> 6, rem = g & 63, mt = rem >> 5, lane = rem & 31;
    int b  = mt * 16 + (lane & 15);
    int kh = lane >> 4;
    const float* xb = xbar + b * 1024 + kc * 32;
    float f[16];
    #pragma unroll
    for (int j = 0; j < 8; ++j) f[j]     = xb[kh * 8 + j];
    #pragma unroll
    for (int j = 0; j < 8; ++j) f[8 + j] = xb[16 + kh * 8 + j];
    unsigned w[8];
    #pragma unroll
    for (int j = 0; j < 8; ++j) w[j] = pack_bf16(f[2 * j], f[2 * j + 1]);
    uint4* out = (uint4*)(A + (size_t)g * 16);
    out[0] = make_uint4(w[0], w[1], w[2], w[3]);
    out[1] = make_uint4(w[4], w[5], w[6], w[7]);
}

// ---------------------------------------------------------------- phase 0: alpha, beta
__global__ void k_alphabeta(const float* __restrict__ xbar, const float* __restrict__ wA,
                            const float* __restrict__ wB, float* __restrict__ alpha,
                            float* __restrict__ beta) {
    int t = threadIdx.x;                 // 256 threads: b = t>>3, c = t&7
    int b = t >> 3, c = t & 7;
    const float* w  = (c < 4) ? (wA + c) : (wB + (c - 4));
    const float* xb = xbar + b * 1024;
    float s = 0.f;
    for (int d = 0; d < 1024; ++d) s += xb[d] * w[d * 4];
    float r = 1.0f / (1.0f + __expf(-s)) + 0.001f;
    if (c < 4) alpha[b * 4 + c] = r;
    else       beta[b * 4 + (c - 4)] = r;
}

__global__ void k_init(float* __restrict__ sel) {
    for (int i = threadIdx.x; i < 14 * 32 * 16; i += blockDim.x) sel[i] = 0.f;
}

// ---------------------------------------------------------------- phase 1: GEMM + fused epilogue
// 512 WGs x 256 threads (8 waves).  N-tile = 256 cols/WG, M = 32 (2 WMMA M-tiles).
// Each wave: 2 M-tiles x 2 N-subtiles -> 4 bf16 WMMAs per K-chunk of 32.
__global__ void __launch_bounds__(256) k_gemm(
    const float* __restrict__ wv, const unsigned short* __restrict__ A,
    const int* __restrict__ labels, float* __restrict__ cm_partials,
    float* __restrict__ sel)
{
    __shared__ float          stagF[32 * 256];      // raw f32 W slab (32 KB)
    __shared__ unsigned short Bs[256 * 40];          // bf16 B^T, padded stride 40 (20 KB)
    __shared__ float          contrib[8][32][16];    // per-(wave,lane) cm partials (16 KB)
    __shared__ int            labS[448];

    const int tid    = threadIdx.x;
    const int lane   = tid & 31;
    const int wvid   = tid >> 5;
    const int kh     = lane >> 4;
    const int lanelo = lane & 15;
    const int nbase  = blockIdx.x * 256;

    for (int i = tid; i < 448; i += 256) labS[i] = labels[i];
    __syncthreads();

    v8f acc[2][2];
    #pragma unroll
    for (int mt = 0; mt < 2; ++mt)
        #pragma unroll
        for (int s = 0; s < 2; ++s)
            #pragma unroll
            for (int i = 0; i < 8; ++i) acc[mt][s][i] = 0.0f;

    for (int kc = 0; kc < 32; ++kc) {
        // ---- stage 32x256 f32 of w_vocab into LDS (async DMA path on gfx1250)
        #pragma unroll
        for (int q = 0; q < 8; ++q) {
            int idx = tid + 256 * q;               // 0..2047 float4 items
            int rr  = idx >> 6;                    // k row 0..31
            int cc  = (idx & 63) << 2;             // col*4
            const float* gp = wv + (size_t)(kc * 32 + rr) * 131072 + nbase + cc;
#if USE_ASYNC
            __builtin_amdgcn_global_load_async_to_lds_b128(
                (AS1 v4i_native*)const_cast<float*>(gp),
                (AS3 v4i_native*)&stagF[idx * 4], 0, 0);
#else
            *(float4*)&stagF[idx * 4] = *(const float4*)gp;
#endif
        }
#if USE_ASYNC
#if __has_builtin(__builtin_amdgcn_s_wait_asynccnt)
        __builtin_amdgcn_s_wait_asynccnt(0);
#else
        asm volatile("s_wait_asynccnt 0" ::: "memory");
#endif
#endif
        __syncthreads();

        // ---- transpose + f32->bf16: thread owns column n = tid
        {
            int n = tid;
            unsigned wpk[16];
            #pragma unroll
            for (int k2 = 0; k2 < 16; ++k2)
                wpk[k2] = pack_bf16(stagF[(2 * k2) * 256 + n], stagF[(2 * k2 + 1) * 256 + n]);
            uint4* dst = (uint4*)&Bs[n * 40];
            dst[0] = make_uint4(wpk[0],  wpk[1],  wpk[2],  wpk[3]);
            dst[1] = make_uint4(wpk[4],  wpk[5],  wpk[6],  wpk[7]);
            dst[2] = make_uint4(wpk[8],  wpk[9],  wpk[10], wpk[11]);
            dst[3] = make_uint4(wpk[12], wpk[13], wpk[14], wpk[15]);
        }
        __syncthreads();

        // ---- fragments + 4 WMMAs
        Frag32B af[2], bfr[2];
        #pragma unroll
        for (int mt = 0; mt < 2; ++mt) {
            const uint4* pa = (const uint4*)(A + ((size_t)(kc * 2 + mt) * 32 + lane) * 16);
            af[mt].u.lo = pa[0]; af[mt].u.hi = pa[1];
        }
        #pragma unroll
        for (int s = 0; s < 2; ++s) {
            int nloc = (2 * wvid + s) * 16 + lanelo;
            const uint4* pb = (const uint4*)&Bs[nloc * 40 + kh * 8];
            bfr[s].u.lo = pb[0]; bfr[s].u.hi = pb[2];
        }
        #pragma unroll
        for (int mt = 0; mt < 2; ++mt)
            #pragma unroll
            for (int s = 0; s < 2; ++s)
                acc[mt][s] = __builtin_amdgcn_wmma_f32_16x16x32_bf16(
                    false, af[mt].v, false, bfr[s].v, (short)0, acc[mt][s], false, false);
        __syncthreads();
    }

    // ---- epilogue: core = 0.01*sigmoid(acc)+0.001 ; cm partial + sel gather
    const int r1 = nbase >> 15;                  // constant within a WG (256 | 32768)
    float loc[16];
    #pragma unroll
    for (int i = 0; i < 16; ++i) loc[i] = 0.f;

    #pragma unroll
    for (int mt = 0; mt < 2; ++mt)
        #pragma unroll
        for (int s = 0; s < 2; ++s) {
            int n  = nbase + (2 * wvid + s) * 16 + lanelo;
            int v  = (n >> 2) & 8191;
            int r2 = n & 3;
            #pragma unroll
            for (int r = 0; r < 8; ++r) {
                int b = mt * 16 + r + kh * 8;
                float val = 0.01f / (1.0f + __expf(-acc[mt][s][r])) + 0.001f;
                loc[mt * 8 + r] += val;
                for (int tt = 0; tt < 14; ++tt)
                    if (labS[b * 14 + tt] == v)
                        sel[(tt * 32 + b) * 16 + r1 * 4 + r2] = val;   // unique writer
            }
        }

    #pragma unroll
    for (int e = 0; e < 16; ++e) contrib[wvid][lane][e] = loc[e];
    __syncthreads();

    // fixed-order deterministic reduction -> 128 entries (b, r2) for this WG's r1
    if (tid < 128) {
        int b = tid >> 2, r2 = tid & 3;
        int mt = b >> 4, rem = b & 15;
        int khh = rem >> 3, r = rem & 7;
        int e = mt * 8 + r;
        float ssum = 0.f;
        #pragma unroll
        for (int w = 0; w < 8; ++w)
            #pragma unroll
            for (int q = 0; q < 4; ++q)
                ssum += contrib[w][khh * 16 + r2 + 4 * q][e];
        cm_partials[blockIdx.x * 128 + tid] = ssum;
    }
}

// ---------------------------------------------------------------- phase 2: chains + loss
__device__ __forceinline__ void mm4(float* __restrict__ D, const float* __restrict__ X,
                                    const float* __restrict__ Y) {
    #pragma unroll
    for (int i = 0; i < 4; ++i)
        #pragma unroll
        for (int j = 0; j < 4; ++j) {
            float s = 0.f;
            #pragma unroll
            for (int k = 0; k < 4; ++k) s += X[i * 4 + k] * Y[k * 4 + j];
            D[i * 4 + j] = s;
        }
}

__global__ void k_finalize(const float* __restrict__ alpha, const float* __restrict__ beta,
                           const float* __restrict__ cmp, const float* __restrict__ sel,
                           float* __restrict__ out) {
    int b = threadIdx.x;          // 32 lanes = 1 wave
    float a[4], be[4], M[16], C[16], P[16], T0[16];
    #pragma unroll
    for (int i = 0; i < 4; ++i) { a[i] = alpha[b * 4 + i]; be[i] = beta[b * 4 + i]; }
    // deterministic cm reduction: blocks i*128..i*128+127 hold r1 == i
    #pragma unroll
    for (int i = 0; i < 4; ++i)
        #pragma unroll
        for (int j = 0; j < 4; ++j) {
            float ssum = 0.f;
            for (int blk = i * 128; blk < (i + 1) * 128; ++blk)
                ssum += cmp[blk * 128 + b * 4 + j];
            M[i * 4 + j] = ssum;
        }
    #pragma unroll
    for (int i = 0; i < 16; ++i) C[i] = sel[(0 * 32 + b) * 16 + i];
    for (int t = 1; t < 14; ++t) {
        float S[16];
        #pragma unroll
        for (int i = 0; i < 16; ++i) S[i] = sel[(t * 32 + b) * 16 + i];
        mm4(T0, C, S);
        #pragma unroll
        for (int i = 0; i < 16; ++i) C[i] = T0[i];
    }
    float u = 0.f;
    #pragma unroll
    for (int i = 0; i < 4; ++i)
        #pragma unroll
        for (int j = 0; j < 4; ++j) u += a[i] * C[i * 4 + j] * be[j];
    #pragma unroll
    for (int i = 0; i < 16; ++i) P[i] = M[i];
    for (int t = 0; t < 14; ++t) {            // P = cm^(T+1)
        mm4(T0, P, M);
        #pragma unroll
        for (int i = 0; i < 16; ++i) P[i] = T0[i];
    }
    float nrm = 0.f;
    #pragma unroll
    for (int i = 0; i < 4; ++i)
        #pragma unroll
        for (int j = 0; j < 4; ++j) nrm += a[i] * P[i * 4 + j] * be[j];
    float loss = -logf(u / nrm + 0.001f);
    for (int off = 16; off > 0; off >>= 1) loss += __shfl_down(loss, off);
    if (b == 0) out[0] = loss * (1.0f / 32.0f);
}

// ---------------------------------------------------------------- launcher
extern "C" void kernel_launch(void* const* d_in, const int* in_sizes, int n_in,
                              void* d_out, int out_size, void* d_ws, size_t ws_size,
                              hipStream_t stream) {
    const float* x   = (const float*)d_in[0];   // (32,14,1024)
    const int*   lab = (const int*)d_in[1];     // (32,14)
    const float* wA  = (const float*)d_in[2];   // (1024,4)
    const float* wB  = (const float*)d_in[3];   // (1024,4)
    const float* wv  = (const float*)d_in[4];   // (1024,131072)
    (void)in_sizes; (void)n_in; (void)out_size; (void)ws_size;

    char* ws = (char*)d_ws;
    unsigned short* A_ws   = (unsigned short*)(ws + 0);        //  64 KB
    float*          xbar   = (float*)(ws + 65536);             // 128 KB
    float*          sel    = (float*)(ws + 196608);            //  28 KB
    float*          alpha  = (float*)(ws + 225280);            // 512 B
    float*          beta   = (float*)(ws + 225792);            // 512 B
    float*          cmpart = (float*)(ws + 226304);            // 256 KB

    hipLaunchKernelGGL(k_xbar,      dim3(32),  dim3(256), 0, stream, x, xbar);
    hipLaunchKernelGGL(k_swizzleA,  dim3(8),   dim3(256), 0, stream, xbar, A_ws);
    hipLaunchKernelGGL(k_alphabeta, dim3(1),   dim3(256), 0, stream, xbar, wA, wB, alpha, beta);
    hipLaunchKernelGGL(k_init,      dim3(1),   dim3(256), 0, stream, sel);
    hipLaunchKernelGGL(k_gemm,      dim3(512), dim3(256), 0, stream, wv, A_ws, lab, cmpart, sel);
    hipLaunchKernelGGL(k_finalize,  dim3(1),   dim3(32),  0, stream, alpha, beta, cmpart, sel,
                       (float*)d_out);
}
